// ExpertParallelMoE_34067680592373
// MI455X (gfx1250) — matrix operand order repaired and checked by reading, not compiled, not run on hardware
//
#include <hip/hip_runtime.h>
#include <hip/hip_bf16.h>
#include <math.h>
#include <stdint.h>

// ---------------- problem constants ----------------
#define T_TOK 2048
#define H_DIM 2048
#define I_DIM 1408
#define N_EXP 8
#define TOPK  2
#define NGRP  4
#define QMAXF 127.0f

#define USE_ASYNC_LDS 1

typedef __bf16 bf16x16 __attribute__((ext_vector_type(16)));
typedef __bf16 bf16x8  __attribute__((ext_vector_type(8)));
typedef float  v8f     __attribute__((ext_vector_type(8)));

// ---- CDNA5 async global->LDS copy (ASYNCcnt-tracked, bypasses VGPRs) ----
__device__ __forceinline__ void async_copy_b128(uint32_t lds_addr,
                                                uint64_t gaddr) {
  asm volatile("global_load_async_to_lds_b128 %0, %1, off"
               :
               : "v"(lds_addr), "v"(gaddr)
               : "memory");
}
__device__ __forceinline__ void async_wait0() {
  asm volatile("s_wait_asynccnt 0" ::: "memory");
}

// =====================================================================
// K0: zero d_out + routing accumulators
// =====================================================================
__global__ __launch_bounds__(256)
void k_zero(float* __restrict__ out, size_t n, int* __restrict__ ec,
            float* __restrict__ sumn) {
  size_t i = (size_t)blockIdx.x * 256 + threadIdx.x;
  if (i < n) out[i] = 0.f;
  if (blockIdx.x == 0 && threadIdx.x < N_EXP) {
    ec[threadIdx.x]   = 0;
    sumn[threadIdx.x] = 0.f;
  }
}

// =====================================================================
// K1: router — one block per token
// =====================================================================
__global__ __launch_bounds__(256)
void k_router(const float* __restrict__ tokens, const float* __restrict__ wr,
              int* __restrict__ ec, int* __restrict__ etok,
              float* __restrict__ ewt, float* __restrict__ sumn) {
  const int t   = blockIdx.x;
  const int tid = threadIdx.x;
  const float* tk = tokens + (size_t)t * H_DIM;

  float p[N_EXP];
#pragma unroll
  for (int e = 0; e < N_EXP; ++e) p[e] = 0.f;

  for (int h = tid; h < H_DIM; h += 256) {
    float x = tk[h];
    v8f wrow = *(const v8f*)(wr + (size_t)h * N_EXP);
#pragma unroll
    for (int e = 0; e < N_EXP; ++e) p[e] += x * wrow[e];
  }
  // wave32 reduction
#pragma unroll
  for (int off = 1; off < 32; off <<= 1) {
#pragma unroll
    for (int e = 0; e < N_EXP; ++e) p[e] += __shfl_xor(p[e], off, 32);
  }
  __shared__ float red[8][N_EXP];
  const int wv = tid >> 5, lane = tid & 31;
  if (lane == 0) {
#pragma unroll
    for (int e = 0; e < N_EXP; ++e) red[wv][e] = p[e];
  }
  __syncthreads();
  if (tid == 0) {
    float s[N_EXP];
#pragma unroll
    for (int e = 0; e < N_EXP; ++e) {
      float l = 0.f;
#pragma unroll
      for (int w = 0; w < 8; ++w) l += red[w][e];
      s[e] = 1.f / (1.f + __expf(-l));          // sigmoid
    }
    // group scores (epg = 2)
    float gs[NGRP];
#pragma unroll
    for (int g = 0; g < NGRP; ++g) gs[g] = fmaxf(s[2 * g], s[2 * g + 1]);
    int g0 = 0;
#pragma unroll
    for (int g = 1; g < NGRP; ++g) if (gs[g] > gs[g0]) g0 = g;
    int g1 = -1;
#pragma unroll
    for (int g = 0; g < NGRP; ++g)
      if (g != g0 && (g1 < 0 || gs[g] > gs[g1])) g1 = g;
    float routed[N_EXP];
#pragma unroll
    for (int e = 0; e < N_EXP; ++e) {
      int g = e >> 1;
      routed[e] = (g == g0 || g == g1) ? s[e] : 0.f;
    }
    int i0 = 0;
#pragma unroll
    for (int e = 1; e < N_EXP; ++e) if (routed[e] > routed[i0]) i0 = e;
    int i1 = -1;
#pragma unroll
    for (int e = 0; e < N_EXP; ++e)
      if (e != i0 && (i1 < 0 || routed[e] > routed[i1])) i1 = e;
    float sw = fmaxf(routed[i0] + routed[i1], 1e-9f);
    float w0 = routed[i0] / sw, w1 = routed[i1] / sw;  // SCALING = 1.0
    // aux: normalized score accumulation
    float ssum = 0.f;
#pragma unroll
    for (int e = 0; e < N_EXP; ++e) ssum += s[e];
    ssum = fmaxf(ssum, 1e-9f);
#pragma unroll
    for (int e = 0; e < N_EXP; ++e) atomicAdd(&sumn[e], s[e] / ssum);
    // slot assignment
    int s0 = atomicAdd(&ec[i0], 1);
    etok[i0 * T_TOK + s0] = t; ewt[i0 * T_TOK + s0] = w0;
    int s1 = atomicAdd(&ec[i1], 1);
    etok[i1 * T_TOK + s1] = t; ewt[i1 * T_TOK + s1] = w1;
  }
}

// =====================================================================
// K2: per-1x128-tile fake quant of tokens -> bf16 dispatch payload
// one wave handles one 128-wide tile (4 elems/lane)
// =====================================================================
__global__ __launch_bounds__(256)
void k_quant_tokens(const float* __restrict__ tokens, __bf16* __restrict__ xq) {
  size_t base = ((size_t)blockIdx.x * 256 + threadIdx.x) * 4;
  float4 v = *(const float4*)(tokens + base);
  float amax = fmaxf(fmaxf(fabsf(v.x), fabsf(v.y)),
                     fmaxf(fabsf(v.z), fabsf(v.w)));
#pragma unroll
  for (int off = 1; off < 32; off <<= 1)
    amax = fmaxf(amax, __shfl_xor(amax, off, 32));
  float scale = (amax > 0.f && __builtin_isfinite(amax)) ? amax * (1.f / QMAXF)
                                                         : 1.f;
  float inv = 1.f / scale;
  float q0 = fminf(fmaxf(rintf(v.x * inv), -QMAXF), QMAXF) * scale;
  float q1 = fminf(fmaxf(rintf(v.y * inv), -QMAXF), QMAXF) * scale;
  float q2 = fminf(fmaxf(rintf(v.z * inv), -QMAXF), QMAXF) * scale;
  float q3 = fminf(fmaxf(rintf(v.w * inv), -QMAXF), QMAXF) * scale;
  xq[base + 0] = (__bf16)q0; xq[base + 1] = (__bf16)q1;
  xq[base + 2] = (__bf16)q2; xq[base + 3] = (__bf16)q3;
}

// =====================================================================
// K3: exclusive scan of expert counts (row compaction base offsets)
// =====================================================================
__global__ void k_scan(const int* __restrict__ ec, int* __restrict__ ro) {
  if (blockIdx.x == 0 && threadIdx.x == 0) {
    int a = 0;
#pragma unroll
    for (int e = 0; e < N_EXP; ++e) { ro[e] = a; a += ec[e]; }
    ro[N_EXP] = a;
  }
}

// ---------------- shared WMMA fragment helpers ----------------
// A tile in LDS: row-major [128][32] bf16. ISA A layout (16-bit 16x32):
//   lanes 0-15:  M=lane,    K = {0..7, 16..23}
//   lanes 16-31: M=lane-16, K = {8..15, 24..31}
__device__ __forceinline__ bf16x16 load_a_frag(const __bf16* lA, int row,
                                               int sel) {
  const __bf16* pa = lA + row * 32 + sel * 8;
  bf16x8 lo = *(const bf16x8*)pa;
  bf16x8 hi = *(const bf16x8*)(pa + 16);
  bf16x16 a;
#pragma unroll
  for (int i = 0; i < 8; ++i) { a[i] = lo[i]; a[8 + i] = hi[i]; }
  return a;
}
// B tile in LDS: K-contiguous per column [128 cols][32 ks] bf16.
//   lanes 0-15:  N=lane,    K = {0..7, 16..23} ; lanes 16-31: N=lane-16, K = {8..15, 24..31}
__device__ __forceinline__ bf16x16 load_b_frag(const __bf16* lB, int col,
                                               int sel) {
  const __bf16* pb = lB + col * 32 + sel * 8;
  bf16x8 lo = *(const bf16x8*)pb;
  bf16x8 hi = *(const bf16x8*)(pb + 16);
  bf16x16 b;
#pragma unroll
  for (int i = 0; i < 8; ++i) { b[i] = lo[i]; b[8 + i] = hi[i]; }
  return b;
}

// =====================================================================
// K4: GEMM1 — gathered xq rows x (w_gate | w_up), fused SiLU*up -> gated bf16
// grid.x = E*16 (m-tiles of 128), grid.y = I/128. 256 thr = 8 waves,
// wave w owns M-strip [16w,16w+16) x 128 cols.
// =====================================================================
__global__ __launch_bounds__(256)
void k_gemm_gateup(const __bf16* __restrict__ xq, const float* __restrict__ wg,
                   const float* __restrict__ wu, const int* __restrict__ ec,
                   const int* __restrict__ ro, const int* __restrict__ etok,
                   __bf16* __restrict__ gated) {
  const int e  = blockIdx.x >> 4;
  const int m0 = (blockIdx.x & 15) * 128;
  const int count = ec[e];
  if (m0 >= count) return;
  const int n0   = blockIdx.y * 128;
  const int base = ro[e];

  __shared__ alignas(64) __bf16 lA[128 * 32];
  __shared__ alignas(64) __bf16 lG[128 * 32];
  __shared__ alignas(64) __bf16 lU[128 * 32];

  const int tid = threadIdx.x;
  const int wv = tid >> 5, lane = tid & 31;
  const int hrow = lane & 15, sel = lane >> 4;

  v8f accg[8] = {}; v8f accu[8] = {};

  const float* wgE = wg + (size_t)e * H_DIM * I_DIM;
  const float* wuE = wu + (size_t)e * H_DIM * I_DIM;

  const int arow = tid >> 1, ahalf = tid & 1;
  int tok = -1;
  if (m0 + arow < count) tok = etok[e * T_TOK + m0 + arow];
  const uint32_t lA_dst =
      (uint32_t)(uintptr_t)(lA + arow * 32 + ahalf * 16);

  for (int k0 = 0; k0 < H_DIM; k0 += 32) {
    // stage A (gathered, zero-padded): async DMA global->LDS for valid rows
    if (tok >= 0) {
#if USE_ASYNC_LDS
      uint64_t ga =
          (uint64_t)(uintptr_t)(xq + (size_t)tok * H_DIM + k0 + ahalf * 16);
      async_copy_b128(lA_dst, ga);
      async_copy_b128(lA_dst + 16, ga + 16);
#else
      *(bf16x16*)(lA + arow * 32 + ahalf * 16) =
          *(const bf16x16*)(xq + (size_t)tok * H_DIM + k0 + ahalf * 16);
#endif
    } else {
      bf16x16 z = {};
      *(bf16x16*)(lA + arow * 32 + ahalf * 16) = z;
    }
    // stage B (f32 -> bf16, K-contiguous per column)
#pragma unroll
    for (int j = 0; j < 16; ++j) {
      int idx = tid + j * 256;
      int col = idx & 127, kk = idx >> 7;
      size_t go = (size_t)(k0 + kk) * I_DIM + n0 + col;
      lG[col * 32 + kk] = (__bf16)wgE[go];
      lU[col * 32 + kk] = (__bf16)wuE[go];
    }
    if (k0 + 32 < H_DIM) {
      __builtin_prefetch(wgE + (size_t)(k0 + 32) * I_DIM + n0 + (tid & 127), 0, 1);
      __builtin_prefetch(wuE + (size_t)(k0 + 32) * I_DIM + n0 + (tid & 127), 0, 1);
    }
#if USE_ASYNC_LDS
    async_wait0();
#endif
    __syncthreads();
    bf16x16 a = load_a_frag(lA, wv * 16 + hrow, sel);
#pragma unroll
    for (int nn = 0; nn < 8; ++nn) {
      bf16x16 bg = load_b_frag(lG, nn * 16 + hrow, sel);
      accg[nn] = __builtin_amdgcn_wmma_f32_16x16x32_bf16(
          false, a, false, bg, (short)0, accg[nn], false, false);
      bf16x16 bu = load_b_frag(lU, nn * 16 + hrow, sel);
      accu[nn] = __builtin_amdgcn_wmma_f32_16x16x32_bf16(
          false, a, false, bu, (short)0, accu[nn], false, false);
    }
    __syncthreads();
  }
  // epilogue: silu(gate)*up, store bf16 (compacted rows)
#pragma unroll
  for (int nn = 0; nn < 8; ++nn) {
    int col = n0 + nn * 16 + hrow;
#pragma unroll
    for (int r = 0; r < 8; ++r) {
      int ml = wv * 16 + sel * 8 + r;     // C/D layout: lanes>=16 hold M=8+r
      if (m0 + ml < count) {
        float g = accg[nn][r], u = accu[nn][r];
        float res = g / (1.f + __expf(-g)) * u;
        gated[(size_t)(base + m0 + ml) * I_DIM + col] = (__bf16)res;
      }
    }
  }
}

// =====================================================================
// K5: GEMM2 — gated x w_down, fused per-row-per-128 fake-quant +
// weighted atomic scatter into out. N-tile (128) == quant tile (128).
// =====================================================================
__global__ __launch_bounds__(256)
void k_gemm_down(const __bf16* __restrict__ gated, const float* __restrict__ wd,
                 const int* __restrict__ ec, const int* __restrict__ ro,
                 const int* __restrict__ etok, const float* __restrict__ ewt,
                 float* __restrict__ out) {
  const int e  = blockIdx.x >> 4;
  const int m0 = (blockIdx.x & 15) * 128;
  const int count = ec[e];
  if (m0 >= count) return;
  const int n0   = blockIdx.y * 128;
  const int base = ro[e];

  __shared__ alignas(64) __bf16 lA[128 * 32];
  __shared__ alignas(64) __bf16 lB[128 * 32];

  const int tid = threadIdx.x;
  const int wv = tid >> 5, lane = tid & 31;
  const int hrow = lane & 15, sel = lane >> 4;

  v8f acc[8] = {};
  const float* wdE = wd + (size_t)e * I_DIM * H_DIM;
  const int arow = tid >> 1, ahalf = tid & 1;
  const bool arow_ok = (m0 + arow) < count;
  const uint32_t lA_dst =
      (uint32_t)(uintptr_t)(lA + arow * 32 + ahalf * 16);
  const __bf16* gsrc =
      gated + (size_t)(base + m0 + arow) * I_DIM + ahalf * 16;

  for (int k0 = 0; k0 < I_DIM; k0 += 32) {
    if (arow_ok) {
#if USE_ASYNC_LDS
      uint64_t ga = (uint64_t)(uintptr_t)(gsrc + k0);
      async_copy_b128(lA_dst, ga);
      async_copy_b128(lA_dst + 16, ga + 16);
#else
      *(bf16x16*)(lA + arow * 32 + ahalf * 16) = *(const bf16x16*)(gsrc + k0);
#endif
    } else {
      bf16x16 z = {};
      *(bf16x16*)(lA + arow * 32 + ahalf * 16) = z;
    }
#pragma unroll
    for (int j = 0; j < 16; ++j) {
      int idx = tid + j * 256;
      int col = idx & 127, kk = idx >> 7;
      lB[col * 32 + kk] = (__bf16)wdE[(size_t)(k0 + kk) * H_DIM + n0 + col];
    }
    if (k0 + 32 < I_DIM)
      __builtin_prefetch(wdE + (size_t)(k0 + 32) * H_DIM + n0 + (tid & 127), 0, 1);
#if USE_ASYNC_LDS
    async_wait0();
#endif
    __syncthreads();
    bf16x16 a = load_a_frag(lA, wv * 16 + hrow, sel);
#pragma unroll
    for (int nn = 0; nn < 8; ++nn) {
      bf16x16 b = load_b_frag(lB, nn * 16 + hrow, sel);
      acc[nn] = __builtin_amdgcn_wmma_f32_16x16x32_bf16(
          false, a, false, b, (short)0, acc[nn], false, false);
    }
    __syncthreads();
  }
  // epilogue: per output row, absmax over this 128-wide quant tile
  // (8 nn-values per lane x 16 lanes of the same half = full row segment)
#pragma unroll
  for (int r = 0; r < 8; ++r) {
    int ml = wv * 16 + sel * 8 + r;
    float amax = 0.f;
#pragma unroll
    for (int nn = 0; nn < 8; ++nn) amax = fmaxf(amax, fabsf(acc[nn][r]));
#pragma unroll
    for (int off = 1; off < 16; off <<= 1)   // xor {1,2,4,8}: stays in half
      amax = fmaxf(amax, __shfl_xor(amax, off, 32));
    float scale = (amax > 0.f && __builtin_isfinite(amax))
                      ? amax * (1.f / QMAXF) : 1.f;
    float inv = 1.f / scale;
    if (m0 + ml < count) {
      int   t = etok[e * T_TOK + m0 + ml];
      float w = ewt[e * T_TOK + m0 + ml];
#pragma unroll
      for (int nn = 0; nn < 8; ++nn) {
        float q = fminf(fmaxf(rintf(acc[nn][r] * inv), -QMAXF), QMAXF) * scale;
        atomicAdd(out + (size_t)t * H_DIM + n0 + nn * 16 + hrow, w * q);
      }
    }
  }
}

// =====================================================================
// K6: aux loss finalize
// =====================================================================
__global__ void k_aux(const int* __restrict__ ec, const float* __restrict__ sumn,
                      float* __restrict__ out) {
  if (blockIdx.x == 0 && threadIdx.x == 0) {
    float aux = 0.f;
#pragma unroll
    for (int e = 0; e < N_EXP; ++e)
      aux += ((float)ec[e] / (float)(T_TOK * TOPK)) *
             (sumn[e] / (float)T_TOK);
    out[(size_t)T_TOK * H_DIM] = aux * (float)N_EXP;
  }
}

// =====================================================================
extern "C" void kernel_launch(void* const* d_in, const int* in_sizes, int n_in,
                              void* d_out, int out_size, void* d_ws,
                              size_t ws_size, hipStream_t stream) {
  const float* tokens   = (const float*)d_in[0];
  const float* w_router = (const float*)d_in[1];
  const float* w_gate   = (const float*)d_in[2];
  const float* w_up     = (const float*)d_in[3];
  const float* w_down   = (const float*)d_in[4];
  float* out = (float*)d_out;

  // workspace carve (256-byte aligned)
  char* w = (char*)d_ws;
  auto carve = [&](size_t bytes) -> void* {
    void* p = (void*)w;
    w += (bytes + 255) & ~(size_t)255;
    return p;
  };
  __bf16* xq    = (__bf16*)carve((size_t)T_TOK * H_DIM * 2);          // 8 MB
  int*    ec    = (int*)   carve(N_EXP * sizeof(int));
  int*    ro    = (int*)   carve((N_EXP + 1) * sizeof(int));
  float*  sumn  = (float*) carve(N_EXP * sizeof(float));
  int*    etok  = (int*)   carve((size_t)N_EXP * T_TOK * sizeof(int));
  float*  ewt   = (float*) carve((size_t)N_EXP * T_TOK * sizeof(float));
  __bf16* gated = (__bf16*)carve((size_t)TOPK * T_TOK * I_DIM * 2);   // 11.5 MB

  const size_t nout = (size_t)out_size;

  // K0: zero out + counters
  k_zero<<<(unsigned)((nout + 255) / 256), 256, 0, stream>>>(out, nout, ec, sumn);
  // K1: router (one block per token)
  k_router<<<T_TOK, 256, 0, stream>>>(tokens, w_router, ec, etok, ewt, sumn);
  // K2: dispatch-payload fake quant (4096 blocks x 1024 elems)
  k_quant_tokens<<<(T_TOK * H_DIM) / 1024, 256, 0, stream>>>(tokens, xq);
  // K3: compaction offsets
  k_scan<<<1, 32, 0, stream>>>(ec, ro);
  // K4: gate/up GEMM + SiLU  (E*16 m-tiles, 11 n-tiles)
  k_gemm_gateup<<<dim3(N_EXP * 16, I_DIM / 128), 256, 0, stream>>>(
      xq, w_gate, w_up, ec, ro, etok, gated);
  // K5: down GEMM + combine quant + weighted scatter (E*16, 16 n-tiles)
  k_gemm_down<<<dim3(N_EXP * 16, H_DIM / 128), 256, 0, stream>>>(
      gated, w_down, ec, ro, etok, ewt, out);
  // K6: aux loss
  k_aux<<<1, 32, 0, stream>>>(ec, sumn, out);

  (void)in_sizes; (void)n_in; (void)ws_size;
}
